// MultiheadAttention_58652073394807
// MI455X (gfx1250) — compile-verified, matrix operand order checked by reading, not run on hardware
//
#include <hip/hip_runtime.h>
#include <hip/hip_bf16.h>

typedef __attribute__((ext_vector_type(16))) _Float16 v16h;
typedef __attribute__((ext_vector_type(8)))  float    v8f;
typedef __attribute__((ext_vector_type(4)))  unsigned int u32x4;
typedef __attribute__((ext_vector_type(8)))  int      i32x8;
typedef __attribute__((ext_vector_type(4)))  int      i32x4;

#define EMBED    1024
#define HEADS    16
#define HEAD_DIM 64
#define BATCH    2
#define SEQ      2048
#define NTOK     (BATCH * SEQ)   // 4096

#if defined(__has_builtin)
#if __has_builtin(__builtin_amdgcn_tensor_load_to_lds)
#define HAVE_TDM 1
#endif
#endif
#ifndef HAVE_TDM
#define HAVE_TDM 0
#endif

// ---------------------------------------------------------------------------
// WMMA helper: D = A(16x32 f16) x B(32x16 f16) + C(16x16 f32)
// ---------------------------------------------------------------------------
__device__ __forceinline__ v8f wmma16x16x32(v16h a, v16h b, v8f c) {
  return __builtin_amdgcn_wmma_f32_16x16x32_f16(
      /*neg_a=*/false, a, /*neg_b=*/false, b,
      /*c_mod=*/(short)0, c, /*reuse_a=*/false, /*reuse_b=*/false);
}

// ---------------------------------------------------------------------------
// 16-row x 32-K f16 fragment gather from LDS (row-major, stride ldk halves).
// ISA 7.12.2 16-bit A layout; also serves B when the operand is N x K
// row-major (X @ W.T / Q @ K.T pattern). Consecutive K pairs coalesce into
// ds_load_b128s.
// ---------------------------------------------------------------------------
__device__ __forceinline__ v16h load_frag(const _Float16* p, int ldk) {
  const int lane = threadIdx.x & 31;
  const int r = lane & 15;
  const int g = lane >> 4;
  union { v16h v; unsigned int u[8]; } f;
  const unsigned int* row = (const unsigned int*)(p + (size_t)r * ldk);
#pragma unroll
  for (int i = 0; i < 8; ++i) {
    const int kpair = i + 4 * g + ((i >= 4) ? 4 : 0);  // K = 2*kpair
    f.u[i] = row[kpair];
  }
  return f.v;
}

// ---------------------------------------------------------------------------
// DPP8 xor-lane exchange (groups of 8 lanes, pure VALU, no LDS round-trip)
// ---------------------------------------------------------------------------
template <unsigned X>
__device__ __forceinline__ float dpp8_xor(float x) {
  constexpr unsigned sel = ((0u ^ X)) | ((1u ^ X) << 3) | ((2u ^ X) << 6) |
                           ((3u ^ X) << 9) | ((4u ^ X) << 12) |
                           ((5u ^ X) << 15) | ((6u ^ X) << 18) |
                           ((7u ^ X) << 21);
  int r = __builtin_amdgcn_mov_dpp8(__builtin_bit_cast(int, x), sel);
  return __builtin_bit_cast(float, r);
}
__device__ __forceinline__ float red16_max(float x) {
  x = fmaxf(x, dpp8_xor<1>(x));
  x = fmaxf(x, dpp8_xor<2>(x));
  x = fmaxf(x, dpp8_xor<4>(x));
  x = fmaxf(x, __shfl_xor(x, 8, 32));  // crosses the 8-lane group
  return x;
}
__device__ __forceinline__ float red16_sum(float x) {
  x += dpp8_xor<1>(x);
  x += dpp8_xor<2>(x);
  x += dpp8_xor<4>(x);
  x += __shfl_xor(x, 8, 32);
  return x;
}

// ---------------------------------------------------------------------------
// Tensor Data Mover: 2-D f16 tile (tile_x elems contiguous, tile_y rows,
// row stride stride_elems) from global -> LDS. Descriptor per ISA 8.3/8.4.
// Issue from one wave; complete with s_wait_tensorcnt.
// ---------------------------------------------------------------------------
#if HAVE_TDM
__device__ __forceinline__ void tdm_load_2d(unsigned lds_byte_addr,
                                            const void* gptr, int tile_x,
                                            int tile_y, int stride_elems) {
  const unsigned long long ga = (unsigned long long)gptr;
  u32x4 g0;
  g0[0] = 1u;                                        // count=1, user desc
  g0[1] = lds_byte_addr;                             // lds_addr
  g0[2] = (unsigned)(ga & 0xFFFFFFFFu);              // global_addr[31:0]
  g0[3] = (unsigned)((ga >> 32) & 0x01FFFFFFu) | 0x80000000u;  // type=2
  i32x8 g1;
  g1[0] = (int)(1u << 16);                           // data_size=1 (2 bytes)
  g1[1] = (int)(((unsigned)tile_x & 0xFFFFu) << 16); // tensor_dim0[15:0]
  g1[2] = (int)((((unsigned)tile_x >> 16) & 0xFFFFu) |
                (((unsigned)tile_y & 0xFFFFu) << 16));   // dim0 hi | dim1 lo
  g1[3] = (int)((((unsigned)tile_y >> 16) & 0xFFFFu) |
                (((unsigned)tile_x & 0xFFFFu) << 16));   // dim1 hi | tile_dim0
  g1[4] = (int)((unsigned)tile_y & 0xFFFFu);             // tile_dim1
  g1[5] = (int)(unsigned)stride_elems;                   // dim0_stride[31:0]
  g1[6] = 0;
  g1[7] = 0;
  i32x4 z4 = {0, 0, 0, 0};
#if __clang_major__ >= 23
  i32x8 z8 = {0, 0, 0, 0, 0, 0, 0, 0};
  __builtin_amdgcn_tensor_load_to_lds(g0, g1, z4, z4, z8, 0);
#else
  __builtin_amdgcn_tensor_load_to_lds(g0, g1, z4, z4, 0);
#endif
}
__device__ __forceinline__ unsigned lds_off(const void* p) {
  return (unsigned)(unsigned long long)p;  // LDS aperture: addr[31:0]
}
#endif

// ---------------------------------------------------------------------------
// f32 -> f16 cast
// ---------------------------------------------------------------------------
__global__ void cast_f32_to_f16(const float* __restrict__ in,
                                _Float16* __restrict__ out, int n) {
  int i = blockIdx.x * blockDim.x + threadIdx.x;
  if (i < n) out[i] = (_Float16)in[i];
}

// ---------------------------------------------------------------------------
// Out[m, e] = sum_k X[m,k] * W[e,k] + bias[e]      (torch Linear semantics)
// Block tile 128x128, 8 waves; each wave 32x64 via 2x4 WMMA accumulators.
// A/B tiles staged through LDS with TDM when available.
// ---------------------------------------------------------------------------
template <bool OUT_F32>
__global__ __launch_bounds__(256) void gemm_xwT_bias(
    const _Float16* __restrict__ X, const _Float16* __restrict__ W,
    const float* __restrict__ bias, void* __restrict__ out) {
  const int K = EMBED;
  __shared__ __align__(16) _Float16 As[128 * 32];
  __shared__ __align__(16) _Float16 Bs[128 * 32];

  const int tid = threadIdx.x;
  const int w = tid >> 5;
  const int mw = w & 3;   // 4 M-slices of 32
  const int nw = w >> 2;  // 2 N-slices of 64
  const int m0blk = blockIdx.y * 128;
  const int n0blk = blockIdx.x * 128;

  v8f acc[2][4];
#pragma unroll
  for (int a = 0; a < 2; ++a)
#pragma unroll
    for (int b = 0; b < 4; ++b)
#pragma unroll
      for (int i = 0; i < 8; ++i) acc[a][b][i] = 0.0f;

  for (int kb = 0; kb < K; kb += 32) {
    __syncthreads();
#if HAVE_TDM
    if (tid < 32) {  // wave 0 drives the TDM; EXEC-independent scalar DMA
      tdm_load_2d(lds_off(As), &X[(size_t)m0blk * K + kb], 32, 128, K);
      tdm_load_2d(lds_off(Bs), &W[(size_t)n0blk * K + kb], 32, 128, K);
      __builtin_amdgcn_s_wait_tensorcnt(0);
    }
#else
#pragma unroll
    for (int it = 0; it < 2; ++it) {
      int c = tid + it * 256;  // 512 chunks of 8 halves each
      int r = c >> 2;
      int kc = (c & 3) * 8;
      *(uint4*)&As[r * 32 + kc] =
          *(const uint4*)&X[(size_t)(m0blk + r) * K + kb + kc];
      *(uint4*)&Bs[r * 32 + kc] =
          *(const uint4*)&W[(size_t)(n0blk + r) * K + kb + kc];
    }
    if (kb + 32 < K) {
      __builtin_prefetch(&X[(size_t)(m0blk + (tid >> 1)) * K + kb + 32], 0, 1);
      __builtin_prefetch(&W[(size_t)(n0blk + (tid >> 1)) * K + kb + 32], 0, 1);
    }
#endif
    __syncthreads();

    v16h a0 = load_frag(&As[(mw * 32) * 32], 32);
    v16h a1 = load_frag(&As[(mw * 32 + 16) * 32], 32);
    v16h bf[4];
#pragma unroll
    for (int j = 0; j < 4; ++j)
      bf[j] = load_frag(&Bs[(nw * 64 + j * 16) * 32], 32);
#pragma unroll
    for (int j = 0; j < 4; ++j) {
      acc[0][j] = wmma16x16x32(a0, bf[j], acc[0][j]);
      acc[1][j] = wmma16x16x32(a1, bf[j], acc[1][j]);
    }
  }

  const int lane = tid & 31;
  const int n = lane & 15;
  const int g = lane >> 4;
#pragma unroll
  for (int am = 0; am < 2; ++am)
#pragma unroll
    for (int bn = 0; bn < 4; ++bn) {
      const int col = n0blk + nw * 64 + bn * 16 + n;
      const float bv = bias[col];
#pragma unroll
      for (int i = 0; i < 8; ++i) {
        const int row = m0blk + mw * 32 + am * 16 + i + 8 * g;
        const float v = acc[am][bn][i] + bv;
        if (OUT_F32)
          ((float*)out)[(size_t)row * EMBED + col] = v;
        else
          ((_Float16*)out)[(size_t)row * EMBED + col] = (_Float16)v;
      }
    }
}

// ---------------------------------------------------------------------------
// Flash attention: one (batch, head, 64-query-row) block; 4 waves x 16 rows;
// streaming softmax over key chunks of 64. K chunks via TDM; V transposed
// manually into LDS (TDM cannot transpose).
// ---------------------------------------------------------------------------
__global__ __launch_bounds__(128) void flash_attn(
    const _Float16* __restrict__ Q, const _Float16* __restrict__ Km,
    const _Float16* __restrict__ Vm, _Float16* __restrict__ O) {
  __shared__ __align__(16) _Float16 Qs[64 * 64];
  __shared__ __align__(16) _Float16 Ks[64 * 64];     // [key][d]
  __shared__ __align__(16) _Float16 VsT[64 * 64];    // [d][key]
  __shared__ __align__(16) _Float16 Ps[4][16 * 64];  // per-wave P staging

  const int tid = threadIdx.x;
  const int w = tid >> 5;
  const int lane = tid & 31;
  const int n = lane & 15;
  const int g = lane >> 4;

  const int qblocks = SEQ / 64;
  const int bid = blockIdx.x;
  const int b = bid / (HEADS * qblocks);
  const int h = (bid / qblocks) % HEADS;
  const int qc = bid % qblocks;
  const size_t headoff = (size_t)b * SEQ * EMBED + (size_t)h * HEAD_DIM;
  const _Float16* Qh = Q + headoff;
  const _Float16* Kh = Km + headoff;
  const _Float16* Vh = Vm + headoff;
  _Float16* Oh = O + headoff;
  const int qbase = qc * 64;

  // Stage 64x64 Q tile
#if HAVE_TDM
  if (tid < 32) {
    tdm_load_2d(lds_off(Qs), &Qh[(size_t)qbase * EMBED], 64, 64, EMBED);
    __builtin_amdgcn_s_wait_tensorcnt(0);
  }
#else
#pragma unroll
  for (int it = 0; it < 4; ++it) {
    int c = tid + it * 128;
    int r = c >> 3;
    int dc = (c & 7) * 8;
    *(uint4*)&Qs[r * 64 + dc] =
        *(const uint4*)&Qh[(size_t)(qbase + r) * EMBED + dc];
  }
#endif
  __syncthreads();

  const v16h aq0 = load_frag(&Qs[(w * 16) * 64], 64);       // d = 0..31
  const v16h aq1 = load_frag(&Qs[(w * 16) * 64 + 32], 64);  // d = 32..63

  v8f accO[4];
  float m_i[8], l_i[8];
#pragma unroll
  for (int t = 0; t < 4; ++t)
#pragma unroll
    for (int i = 0; i < 8; ++i) accO[t][i] = 0.0f;
#pragma unroll
  for (int i = 0; i < 8; ++i) { m_i[i] = -1e30f; l_i[i] = 0.0f; }

  for (int kc2 = 0; kc2 < SEQ / 64; ++kc2) {
    __syncthreads();  // previous chunk fully consumed
#if HAVE_TDM
    if (tid < 32)
      tdm_load_2d(lds_off(Ks), &Kh[(size_t)(kc2 * 64) * EMBED], 64, 64, EMBED);
#endif
    // V chunk transposed into VsT (and K manually when no TDM)
#pragma unroll
    for (int it = 0; it < 4; ++it) {
      int c = tid + it * 128;
      int r = c >> 3;
      int dc = (c & 7) * 8;
#if !HAVE_TDM
      *(uint4*)&Ks[r * 64 + dc] =
          *(const uint4*)&Kh[(size_t)(kc2 * 64 + r) * EMBED + dc];
#endif
      union { uint4 u; _Float16 hh[8]; } vv;
      vv.u = *(const uint4*)&Vh[(size_t)(kc2 * 64 + r) * EMBED + dc];
#pragma unroll
      for (int j = 0; j < 8; ++j) VsT[(dc + j) * 64 + r] = vv.hh[j];
    }
#if HAVE_TDM
    if (tid < 32) __builtin_amdgcn_s_wait_tensorcnt(0);
#endif
    __syncthreads();

    // S = Q K^T : 16 q rows x 64 keys per wave
    v8f accS[4];
#pragma unroll
    for (int j = 0; j < 4; ++j) {
#pragma unroll
      for (int i = 0; i < 8; ++i) accS[j][i] = 0.0f;
      v16h bk0 = load_frag(&Ks[(j * 16) * 64], 64);       // d 0..31
      v16h bk1 = load_frag(&Ks[(j * 16) * 64 + 32], 64);  // d 32..63
      accS[j] = wmma16x16x32(aq0, bk0, accS[j]);
      accS[j] = wmma16x16x32(aq1, bk1, accS[j]);
    }

    const float scale = 0.125f;  // 1/sqrt(64)
    float rmax[8];
#pragma unroll
    for (int i = 0; i < 8; ++i) {
      float mx = -1e30f;
#pragma unroll
      for (int j = 0; j < 4; ++j) {
        accS[j][i] *= scale;
        mx = fmaxf(mx, accS[j][i]);
      }
      rmax[i] = red16_max(mx);  // DPP8 + one bpermute, per C-layout row group
    }

    float rsum[8];
#pragma unroll
    for (int i = 0; i < 8; ++i) {
      const float mnew = fmaxf(m_i[i], rmax[i]);
      const float alpha = __expf(m_i[i] - mnew);
      m_i[i] = mnew;
      float s = 0.0f;
#pragma unroll
      for (int j = 0; j < 4; ++j) {
        const float p = __expf(accS[j][i] - mnew);
        accS[j][i] = p;
        s += p;
      }
      rsum[i] = red16_sum(s);
      l_i[i] = l_i[i] * alpha + rsum[i];
#pragma unroll
      for (int t = 0; t < 4; ++t) accO[t][i] *= alpha;
    }

    // Spill P (16x64) as f16 to LDS to re-load in A-fragment layout
    _Float16* Pw = &Ps[w][0];
#pragma unroll
    for (int j = 0; j < 4; ++j)
#pragma unroll
      for (int i = 0; i < 8; ++i)
        Pw[(i + 8 * g) * 64 + j * 16 + n] = (_Float16)accS[j][i];
    __syncthreads();

    // O += P V   (A = P rows over keys, B = V columns via VsT rows)
    v16h ap0 = load_frag(Pw, 64);       // keys 0..31
    v16h ap1 = load_frag(Pw + 32, 64);  // keys 32..63
#pragma unroll
    for (int t = 0; t < 4; ++t) {
      v16h bv0 = load_frag(&VsT[(t * 16) * 64], 64);
      v16h bv1 = load_frag(&VsT[(t * 16) * 64 + 32], 64);
      accO[t] = wmma16x16x32(ap0, bv0, accO[t]);
      accO[t] = wmma16x16x32(ap1, bv1, accO[t]);
    }
  }

  // Normalize and write O (f16)
#pragma unroll
  for (int i = 0; i < 8; ++i) {
    const float inv = 1.0f / l_i[i];
    const int qrow = qbase + w * 16 + i + 8 * g;
#pragma unroll
    for (int t = 0; t < 4; ++t)
      Oh[(size_t)qrow * EMBED + t * 16 + n] = (_Float16)(accO[t][i] * inv);
  }
}

// ---------------------------------------------------------------------------
// Host launch
// ---------------------------------------------------------------------------
extern "C" void kernel_launch(void* const* d_in, const int* in_sizes, int n_in,
                              void* d_out, int out_size, void* d_ws,
                              size_t ws_size, hipStream_t stream) {
  const float* query = (const float*)d_in[0];
  const float* key_  = (const float*)d_in[1];
  const float* value = (const float*)d_in[2];
  const float* Wq = (const float*)d_in[3];
  const float* bq = (const float*)d_in[4];
  const float* Wk = (const float*)d_in[5];
  const float* bk = (const float*)d_in[6];
  const float* Wv = (const float*)d_in[7];
  const float* bv = (const float*)d_in[8];
  const float* Wo = (const float*)d_in[9];
  const float* bo = (const float*)d_in[10];

  const size_t ACT = (size_t)NTOK * EMBED;
  const size_t WEL = (size_t)EMBED * EMBED;

  char* ws = (char*)d_ws;
  _Float16* Xq16 = (_Float16*)ws;  ws += ACT * sizeof(_Float16);
  _Float16* Xk16 = (_Float16*)ws;  ws += ACT * sizeof(_Float16);
  _Float16* Xv16 = (_Float16*)ws;  ws += ACT * sizeof(_Float16);
  _Float16* Wq16 = (_Float16*)ws;  ws += WEL * sizeof(_Float16);
  _Float16* Wk16 = (_Float16*)ws;  ws += WEL * sizeof(_Float16);
  _Float16* Wv16 = (_Float16*)ws;  ws += WEL * sizeof(_Float16);
  _Float16* Wo16 = (_Float16*)ws;  ws += WEL * sizeof(_Float16);
  _Float16* Qp   = (_Float16*)ws;  ws += ACT * sizeof(_Float16);
  _Float16* Kp   = (_Float16*)ws;  ws += ACT * sizeof(_Float16);
  _Float16* Vp   = (_Float16*)ws;  ws += ACT * sizeof(_Float16);
  _Float16* Oa   = (_Float16*)ws;  ws += ACT * sizeof(_Float16);

  {
    const int na = (int)ACT, nw = (int)WEL;
    cast_f32_to_f16<<<(na + 255) / 256, 256, 0, stream>>>(query, Xq16, na);
    cast_f32_to_f16<<<(na + 255) / 256, 256, 0, stream>>>(key_,  Xk16, na);
    cast_f32_to_f16<<<(na + 255) / 256, 256, 0, stream>>>(value, Xv16, na);
    cast_f32_to_f16<<<(nw + 255) / 256, 256, 0, stream>>>(Wq, Wq16, nw);
    cast_f32_to_f16<<<(nw + 255) / 256, 256, 0, stream>>>(Wk, Wk16, nw);
    cast_f32_to_f16<<<(nw + 255) / 256, 256, 0, stream>>>(Wv, Wv16, nw);
    cast_f32_to_f16<<<(nw + 255) / 256, 256, 0, stream>>>(Wo, Wo16, nw);
  }

  dim3 ggrid(EMBED / 128, NTOK / 128);
  gemm_xwT_bias<false><<<ggrid, 256, 0, stream>>>(Xq16, Wq16, bq, (void*)Qp);
  gemm_xwT_bias<false><<<ggrid, 256, 0, stream>>>(Xk16, Wk16, bk, (void*)Kp);
  gemm_xwT_bias<false><<<ggrid, 256, 0, stream>>>(Xv16, Wv16, bv, (void*)Vp);

  flash_attn<<<BATCH * HEADS * (SEQ / 64), 128, 0, stream>>>(Qp, Kp, Vp, Oa);

  gemm_xwT_bias<true><<<ggrid, 256, 0, stream>>>(Oa, Wo16, bo, d_out);
}
